// DeeperGNN_88295937671289
// MI455X (gfx1250) — matrix operand order repaired and checked by reading, not compiled, not run on hardware
//
#include <hip/hip_runtime.h>
#include <cstdint>

// ---------------- problem constants ----------------
#define NN 16384      // nodes
#define EE 131072     // edges
#define D_IN 128
#define D_E  64
#define HH 256        // hidden
#define OUTW 16
#define LL 4
#define EPS_MSG 1e-7f
#define EPS_SM  1e-16f
#define EPS_LN  1e-5f

typedef __attribute__((ext_vector_type(16))) _Float16 v16h;
typedef __attribute__((ext_vector_type(8)))  float    v8f;
typedef __attribute__((ext_vector_type(4)))  float    v4f;
typedef __attribute__((ext_vector_type(4)))  int      v4i;

// =====================================================================
// GEMM: C[M,N] = A[M,K] @ W[K,N] + bias[N] (+ R[M,N] if HAS_RES)
// fp32 -> f16 fragments, fp32 accumulation via v_wmma_f32_16x16x32_f16.
// Block = 8 waves = 512(M) x 16(N) tile. The K x 16 weight panel is
// cooperatively staged in LDS *pre-swizzled into the B fragment layout*
// (<=16KB), so each wave reads one fragment per K-step as 32B of LDS
// instead of 16 strided VMEM loads. Each wave owns a 64(M) x 16(N)
// strip: per K-step, all 4 A fragments are loaded into distinct
// registers first, then 4 independent WMMAs issue back-to-back (no
// WAR hazard NOPs; next step's loads overlap the WMMAs).
// Requires K%64==0, N%16==0, M%512==0 (true for all shapes here).
// =====================================================================
template <int K, bool HAS_RES>
__global__ void __launch_bounds__(256)
gemm_wmma_f16(const float* __restrict__ A, const float* __restrict__ W,
              const float* __restrict__ bias, const float* __restrict__ R,
              float* __restrict__ C, int M, int N) {
    constexpr int KSTEPS = K / 32;
    // [ks][lane][i] : lane's 16 B-halves contiguous (32B) per K-step
    __shared__ alignas(32) _Float16 bsh[KSTEPS * 512];

    const int tid   = threadIdx.x;
    const int lane  = tid & 31;
    const int wave  = tid >> 5;
    const int half_ = lane >> 4;     // lane group 0/1
    const int l16   = lane & 15;
    const int n0 = blockIdx.x * 16;
    const int m0 = (blockIdx.y * 8 + wave) * 64;
    const int bcol = n0 + l16;

    // ---- stage B panel (K x 16) into LDS, swizzled to fragment layout ----
    // element (k, n): lane = (k%16>=8 ? 16 : 0) + n, i = k%8 + (k%32>=16 ? 8 : 0)
    {
        const int krow = tid >> 2;           // 0..63
        const int nq   = (tid & 3) * 4;      // 0,4,8,12
#pragma unroll
        for (int j = 0; j < K / 64; ++j) {
            const int k = krow + j * 64;
            const v4f wv = *(const v4f*)(W + (size_t)k * N + n0 + nq);
            _Float16* p = &bsh[(k >> 5) * 512 + ((k & 8) * 2 + nq) * 16 + (k & 7) + ((k & 16) >> 1)];
            p[0]  = (_Float16)wv.x;
            p[16] = (_Float16)wv.y;
            p[32] = (_Float16)wv.z;
            p[48] = (_Float16)wv.w;
        }
    }
    __syncthreads();

    v8f acc[4] = {v8f{}, v8f{}, v8f{}, v8f{}};

    // one base address per wave; all fragment/K-step offsets are
    // compile-time constants folded into load-immediate offsets
    const float* abase = A + (size_t)(m0 + l16) * K + half_ * 8;

#pragma unroll
    for (int ks = 0; ks < KSTEPS; ++ks) {
        const v16h b = *(const v16h*)&bsh[ks * 512 + lane * 16];

        v16h a[4];
#pragma unroll
        for (int mt = 0; mt < 4; ++mt) {
            const v4f* ap4 = (const v4f*)(abase + mt * 16 * K + ks * 32);
            const v4f p0 = ap4[0];   // K + {0..3}
            const v4f p1 = ap4[1];   // K + {4..7}
            const v4f q0 = ap4[4];   // K + {16..19}
            const v4f q1 = ap4[5];   // K + {20..23}
            a[mt][0] = (_Float16)p0.x;  a[mt][1] = (_Float16)p0.y;
            a[mt][2] = (_Float16)p0.z;  a[mt][3] = (_Float16)p0.w;
            a[mt][4] = (_Float16)p1.x;  a[mt][5] = (_Float16)p1.y;
            a[mt][6] = (_Float16)p1.z;  a[mt][7] = (_Float16)p1.w;
            a[mt][8]  = (_Float16)q0.x; a[mt][9]  = (_Float16)q0.y;
            a[mt][10] = (_Float16)q0.z; a[mt][11] = (_Float16)q0.w;
            a[mt][12] = (_Float16)q1.x; a[mt][13] = (_Float16)q1.y;
            a[mt][14] = (_Float16)q1.z; a[mt][15] = (_Float16)q1.w;
        }
#pragma unroll
        for (int mt = 0; mt < 4; ++mt)
            acc[mt] = __builtin_amdgcn_wmma_f32_16x16x32_f16(
                false, a[mt], false, b, (short)0, acc[mt], false, false);
    }

    // D layout: VGPR r -> M = base + r + half_*8, N = bcol
    const float bn = bias[bcol];
#pragma unroll
    for (int mt = 0; mt < 4; ++mt) {
#pragma unroll
        for (int r = 0; r < 8; ++r) {
            const int m = m0 + mt * 16 + r + half_ * 8;
            const size_t idx = (size_t)m * N + bcol;
            float v = acc[mt][r] + bn;
            if (HAS_RES) v += R[idx];
            C[idx] = v;
        }
    }
}

// =====================================================================
// LayerNorm (+ReLU) : one wave per row, cross-lane reduce via shfl_xor
// =====================================================================
template <int WIDTH>
__global__ void __launch_bounds__(256)
ln_relu_kernel(const float* __restrict__ X, const float* __restrict__ g,
               const float* __restrict__ b, float* __restrict__ Y, int nrows) {
    const int lane = threadIdx.x & 31;
    const int wave = threadIdx.x >> 5;
    const int row  = blockIdx.x * 8 + wave;
    if (row >= nrows) return;
    constexpr int PER = WIDTH / 32;
    float v[PER];
    const float* xp = X + (size_t)row * WIDTH;
    float s = 0.f, s2 = 0.f;
#pragma unroll
    for (int i = 0; i < PER; ++i) {
        v[i] = xp[lane + i * 32];
        s  += v[i];
        s2 += v[i] * v[i];
    }
#pragma unroll
    for (int off = 16; off > 0; off >>= 1) {
        s  += __shfl_xor(s,  off, 32);
        s2 += __shfl_xor(s2, off, 32);
    }
    const float mu  = s / (float)WIDTH;
    const float var = s2 / (float)WIDTH - mu * mu;
    const float inv = rsqrtf(var + EPS_LN);
    float* yp = Y + (size_t)row * WIDTH;
#pragma unroll
    for (int i = 0; i < PER; ++i) {
        const int c = lane + i * 32;
        float y = (v[i] - mu) * inv * g[c] + b[c];
        yp[c] = fmaxf(y, 0.f);
    }
}

// =====================================================================
// Segment-softmax buffers init: mx = ordered(-inf), den = num = 0
// (mx stored as monotone int encoding of float: f>=0 -> bits,
//  f<0 -> bits ^ 0x7fffffff; INT_MIN is below every encoded value)
// =====================================================================
__global__ void init_seg_kernel(v4i* __restrict__ mx, v4f* __restrict__ den,
                                v4f* __restrict__ num, int n4) {
    const int i = blockIdx.x * 256 + threadIdx.x;
    if (i < n4) {
        mx[i]  = v4i{(int)0x80000000, (int)0x80000000, (int)0x80000000, (int)0x80000000};
        den[i] = v4f{0.f, 0.f, 0.f, 0.f};
        num[i] = v4f{0.f, 0.f, 0.f, 0.f};
    }
}

__device__ __forceinline__ int   f2o(float f) { int i = __float_as_int(f); return i >= 0 ? i : (i ^ 0x7fffffff); }
__device__ __forceinline__ float o2f(int i)   { return __int_as_float(i >= 0 ? i : (i ^ 0x7fffffff)); }

// =====================================================================
// Edge pass 1: msg = relu(x[src]+e)+eps ; atomic segment max of msg*t
// 64 lanes per edge (float4 lanes), 4 edges per 256-thread block.
// e-rows are touched once per pass -> non-temporal loads (keep WGP$
// for the reused gathered node rows; e stays L2-resident).
// =====================================================================
__global__ void __launch_bounds__(256)
edge_max_kernel(const float* __restrict__ xin, const float* __restrict__ e,
                const int* __restrict__ src, const int* __restrict__ dst,
                const float* __restrict__ tptr, int layer, int* __restrict__ mx) {
    const int edge = blockIdx.x * 4 + (threadIdx.x >> 6);
    const int f    = (threadIdx.x & 63) * 4;
    const float t  = tptr[layer];
    const int s = src[edge], d = dst[edge];
    const v4f ev = __builtin_nontemporal_load((const v4f*)(e + (size_t)edge * HH + f));
    const v4f xv = *(const v4f*)(xin + (size_t)s * HH + f);
    int* mp = mx + (size_t)d * HH + f;
    atomicMax(mp + 0, f2o((fmaxf(xv.x + ev.x, 0.f) + EPS_MSG) * t));
    atomicMax(mp + 1, f2o((fmaxf(xv.y + ev.y, 0.f) + EPS_MSG) * t));
    atomicMax(mp + 2, f2o((fmaxf(xv.z + ev.z, 0.f) + EPS_MSG) * t));
    atomicMax(mp + 3, f2o((fmaxf(xv.w + ev.w, 0.f) + EPS_MSG) * t));
}

// =====================================================================
// Edge pass 2 (fused softmax sums): ex = exp(m - mx[dst]);
//   den[dst] += ex ; num[dst] += msg*ex     (agg = num/(den+eps))
// =====================================================================
__global__ void __launch_bounds__(256)
edge_sum_kernel(const float* __restrict__ xin, const float* __restrict__ e,
                const int* __restrict__ src, const int* __restrict__ dst,
                const float* __restrict__ tptr, int layer, const int* __restrict__ mx,
                float* __restrict__ den, float* __restrict__ num) {
    const int edge = blockIdx.x * 4 + (threadIdx.x >> 6);
    const int f    = (threadIdx.x & 63) * 4;
    const float t  = tptr[layer];
    const int s = src[edge], d = dst[edge];
    const size_t di = (size_t)d * HH + f;
    const v4f ev = __builtin_nontemporal_load((const v4f*)(e + (size_t)edge * HH + f));
    const v4f xv = *(const v4f*)(xin + (size_t)s * HH + f);
    const v4i mv = *(const v4i*)(mx + di);
    float msg, ex;
    msg = fmaxf(xv.x + ev.x, 0.f) + EPS_MSG; ex = __expf(msg * t - o2f(mv.x));
    unsafeAtomicAdd(den + di + 0, ex); unsafeAtomicAdd(num + di + 0, msg * ex);
    msg = fmaxf(xv.y + ev.y, 0.f) + EPS_MSG; ex = __expf(msg * t - o2f(mv.y));
    unsafeAtomicAdd(den + di + 1, ex); unsafeAtomicAdd(num + di + 1, msg * ex);
    msg = fmaxf(xv.z + ev.z, 0.f) + EPS_MSG; ex = __expf(msg * t - o2f(mv.z));
    unsafeAtomicAdd(den + di + 2, ex); unsafeAtomicAdd(num + di + 2, msg * ex);
    msg = fmaxf(xv.w + ev.w, 0.f) + EPS_MSG; ex = __expf(msg * t - o2f(mv.w));
    unsafeAtomicAdd(den + di + 3, ex); unsafeAtomicAdd(num + di + 3, msg * ex);
}

// =====================================================================
// agg + root residual, in place into num: num = num/(den+eps) + xin
// =====================================================================
__global__ void agg_combine_kernel(v4f* __restrict__ num, const v4f* __restrict__ den,
                                   const v4f* __restrict__ xin, int n4) {
    const int i = blockIdx.x * 256 + threadIdx.x;
    if (i < n4) {
        v4f nv = num[i];
        const v4f dv = den[i];
        const v4f xv = xin[i];
        nv.x = nv.x / (dv.x + EPS_SM) + xv.x;
        nv.y = nv.y / (dv.y + EPS_SM) + xv.y;
        nv.z = nv.z / (dv.z + EPS_SM) + xv.z;
        nv.w = nv.w / (dv.w + EPS_SM) + xv.w;
        num[i] = nv;
    }
}

// =====================================================================
// Host-side orchestration
// =====================================================================
static inline void launch_gemm(const float* A, const float* W, const float* bias,
                               const float* R, float* C, int M, int K, int N,
                               hipStream_t stream) {
    dim3 grid(N / 16, M / 512);
    if (R) {
        switch (K) {
        case 256: gemm_wmma_f16<256, true><<<grid, 256, 0, stream>>>(A, W, bias, R, C, M, N); break;
        case 512: gemm_wmma_f16<512, true><<<grid, 256, 0, stream>>>(A, W, bias, R, C, M, N); break;
        }
    } else {
        switch (K) {
        case 64:  gemm_wmma_f16<64,  false><<<grid, 256, 0, stream>>>(A, W, bias, nullptr, C, M, N); break;
        case 128: gemm_wmma_f16<128, false><<<grid, 256, 0, stream>>>(A, W, bias, nullptr, C, M, N); break;
        case 256: gemm_wmma_f16<256, false><<<grid, 256, 0, stream>>>(A, W, bias, nullptr, C, M, N); break;
        case 512: gemm_wmma_f16<512, false><<<grid, 256, 0, stream>>>(A, W, bias, nullptr, C, M, N); break;
        }
    }
}

extern "C" void kernel_launch(void* const* d_in, const int* in_sizes, int n_in,
                              void* d_out, int out_size, void* d_ws, size_t ws_size,
                              hipStream_t stream) {
    const float* x     = (const float*)d_in[0];
    const int*   ei    = (const int*)  d_in[1];
    const float* ea    = (const float*)d_in[2];
    const float* enc_w = (const float*)d_in[3];
    const float* enc_b = (const float*)d_in[4];
    const float* ee_w  = (const float*)d_in[5];
    const float* ee_b  = (const float*)d_in[6];
    const float* w1    = (const float*)d_in[7];
    const float* b1    = (const float*)d_in[8];
    const float* lng   = (const float*)d_in[9];
    const float* lnb   = (const float*)d_in[10];
    const float* w2    = (const float*)d_in[11];
    const float* b2    = (const float*)d_in[12];
    const float* t     = (const float*)d_in[13];
    const float* ng    = (const float*)d_in[14];
    const float* nb    = (const float*)d_in[15];
    const float* lin_w = (const float*)d_in[16];
    const float* lin_b = (const float*)d_in[17];
    float* out = (float*)d_out;

    const int* src = ei;
    const int* dst = ei + EE;

    // ---- workspace carve-out (~256 MB) ----
    char* ws = (char*)d_ws;
    size_t off = 0;
    auto carve = [&](size_t bytes) -> void* {
        void* p = ws + off;
        off += (bytes + 255) & ~(size_t)255;
        return p;
    };
    const size_t NHB = (size_t)NN * HH * sizeof(float);
    float* e      = (float*)carve((size_t)EE * HH * sizeof(float)); // 134 MB (L2-resident)
    float* hA     = (float*)carve(NHB);
    float* hB     = (float*)carve(NHB);
    float* z      = (float*)carve(NHB);
    int*   mx     = (int*)  carve(NHB);
    float* den    = (float*)carve(NHB);
    float* num    = (float*)carve(NHB);
    float* buf512 = (float*)carve((size_t)NN * 2 * HH * sizeof(float));
    (void)ws_size;

    const int nNH = NN * HH;
    const int n4  = nNH / 4;
    const int ewBlocks4 = (n4 + 255) / 256;

    // encoders
    launch_gemm(x,  enc_w, enc_b, nullptr, hA, NN, D_IN, HH, stream);  // h = x@enc_w+b
    launch_gemm(ea, ee_w,  ee_b,  nullptr, e,  EE, D_E,  HH, stream);  // e = ea@ee_w+b

    float* hcur = hA;
    float* hnext = hB;
    for (int layer = 0; layer < LL; ++layer) {
        const float* xin;
        if (layer == 0) {
            xin = hcur;                                   // first layer: conv on h directly
        } else {
            ln_relu_kernel<HH><<<NN / 8, 256, 0, stream>>>(
                hcur, ng + layer * HH, nb + layer * HH, z, NN);
            xin = z;                                      // res+: conv(relu(norm(h)))
        }

        init_seg_kernel<<<ewBlocks4, 256, 0, stream>>>(
            (v4i*)mx, (v4f*)den, (v4f*)num, n4);
        edge_max_kernel<<<EE / 4, 256, 0, stream>>>(xin, e, src, dst, t, layer, mx);
        edge_sum_kernel<<<EE / 4, 256, 0, stream>>>(xin, e, src, dst, t, layer, mx, den, num);
        agg_combine_kernel<<<ewBlocks4, 256, 0, stream>>>(
            (v4f*)num, (const v4f*)den, (const v4f*)xin, n4);

        // MLP: Linear(256->512) -> LN -> ReLU -> Linear(512->256) [+ residual h]
        launch_gemm(num, w1 + (size_t)layer * HH * (2 * HH), b1 + layer * (2 * HH),
                    nullptr, buf512, NN, HH, 2 * HH, stream);
        ln_relu_kernel<2 * HH><<<NN / 8, 256, 0, stream>>>(
            buf512, lng + layer * (2 * HH), lnb + layer * (2 * HH), buf512, NN);
        launch_gemm(buf512, w2 + (size_t)layer * (2 * HH) * HH, b2 + layer * HH,
                    (layer == 0) ? nullptr : hcur, hnext, NN, 2 * HH, HH, stream);

        float* tmp = hcur; hcur = hnext; hnext = tmp;
    }

    // final: relu(LN(h, ng[0], nb[0])) @ lin_w + lin_b
    ln_relu_kernel<HH><<<NN / 8, 256, 0, stream>>>(hcur, ng, nb, z, NN);
    launch_gemm(z, lin_w, lin_b, nullptr, out, NN, HH, OUTW, stream);
}
